// InferencePalettizedConv2d_56418690401092
// MI455X (gfx1250) — matrix operand order repaired and checked by reading, not compiled, not run on hardware
//
#include <hip/hip_runtime.h>

typedef __attribute__((ext_vector_type(16))) __bf16         v16bf;
typedef __attribute__((ext_vector_type(2)))  __bf16         v2bf;
typedef __attribute__((ext_vector_type(8)))  float          v8f;
typedef __attribute__((ext_vector_type(8)))  unsigned short v8u;
typedef __attribute__((ext_vector_type(16))) unsigned short v16u;
typedef __attribute__((ext_vector_type(4)))  unsigned int   v4ui;

#define CIN    64
#define COUT   64
#define H      256
#define W      256
#define HO     254
#define WO     254
#define NB     16
#define KTOT   576            // CIN * 3 * 3
#define NCHUNK 18             // K chunks of 32
#define ASTR   584            // A row stride in bf16 (576 + 8 pad -> distinct banks)
#define BSTR   40             // B row stride in bf16 (32 + 8 pad)
#define NTILE  128            // ow columns per block
#define BBUF   (NTILE * BSTR) // one B buffer, in bf16 elements

// fp32 -> bf16 bits (RNE), scalar (used only in weight dequant, amortized)
__device__ __forceinline__ unsigned short f2bfu(float f) {
  unsigned u = __float_as_uint(f);
  u += 0x7FFFu + ((u >> 16) & 1u);
  return (unsigned short)(u >> 16);
}

// pack two fp32 into packed bf16x2 (one dword), RNE.
// Tier 1: native packed cvt if the toolchain declares it.
// Tier 2: 2x(bfe+add3) rounding + one v_perm_b32 to merge the high halves.
__device__ __forceinline__ unsigned int pk_bf16(float a, float b) {
#if __has_builtin(__builtin_amdgcn_cvt_pk_bf16_f32)
  union { v2bf v; unsigned int u; } cv;
  cv.v = __builtin_amdgcn_cvt_pk_bf16_f32(a, b);
  return cv.u;
#else
  unsigned ua = __float_as_uint(a);
  unsigned ub = __float_as_uint(b);
  ua += 0x7FFFu + ((ua >> 16) & 1u);   // v_bfe_u32 + v_add3_u32
  ub += 0x7FFFu + ((ub >> 16) & 1u);
  // dst = { ub[31:16], ua[31:16] }  (sel codes 0-3 -> src1, 4-7 -> src0)
  return __builtin_amdgcn_perm(ub, ua, 0x07060302u);   // v_perm_b32
#endif
}

// 16-element bf16 fragment from two 16B LDS loads (ds_load_b128 x2)
__device__ __forceinline__ v16bf ld_frag(const unsigned short* p0,
                                         const unsigned short* p1) {
  v8u lo = *(const v8u*)p0;
  v8u hi = *(const v8u*)p1;
  union { v16u u; v16bf b; } cv;
#pragma unroll
  for (int j = 0; j < 8; ++j) { cv.u[j] = lo[j]; cv.u[j + 8] = hi[j]; }
  return cv.b;
}

// global pointer for K-chunk c (c = khw*2 + cig) for this staging thread
__device__ __forceinline__ const float* chunk_ptr(const float* x, int bn, int oh,
                                                  int ow0, int scol, int sg, int c) {
  const int khw = c >> 1;
  const int cig = c & 1;
  const int kh  = khw / 3;
  const int kw  = khw - kh * 3;
  int wq = ow0 + scol + kw;
  if (wq > W - 1) wq = W - 1;                 // clamp; outputs masked at store
  const int ci0 = cig * 32 + sg * 16;
  return x + ((size_t)(bn * CIN + ci0) * H + (size_t)(oh + kh)) * W + (size_t)wq;
}

__global__ __launch_bounds__(256)
void palet_conv2d_wmma_bf16(const float* __restrict__ x,
                            const float* __restrict__ lut,
                            const int*   __restrict__ widx,
                            const float* __restrict__ bias,
                            float* __restrict__ out) {
  __shared__ unsigned short A_lds[COUT * ASTR];   // weights bf16, [o][khw*64+ci]
  __shared__ unsigned short B_lds[2 * BBUF];      // im2col bf16, double-buffered [n][k]
  __shared__ float lut_s[256];
  __shared__ float bias_s[COUT];

  const int tid  = threadIdx.x;
  const int lane = tid & 31;
  const int wv   = tid >> 5;      // wave 0..7
  const int mt   = wv & 3;        // M tile (16 couts)
  const int wg   = wv >> 2;       // N group (64 cols)
  const int half = lane >> 4;     // lane half for WMMA layouts
  const int lm   = lane & 15;

  const int b   = blockIdx.x;
  const int bn  = b / (HO * 2);
  const int rr  = b - bn * (HO * 2);
  const int oh  = rr >> 1;
  const int ow0 = (rr & 1) * NTILE;

  const int scol = tid & 127;     // staging: output column within tile
  const int sg   = tid >> 7;      // staging: k half (16 channels each)

  // ---- stage palette + bias ----
  lut_s[tid] = lut[tid];
  if (tid < COUT) bias_s[tid] = bias[tid];
  __syncthreads();

  // ---- dequantize weights into LDS bf16, K reordered to (kh*3+kw)*64 + ci ----
  for (int idx = tid; idx < COUT * KTOT; idx += 256) {
    const int o   = idx / KTOT;
    const int r   = idx - o * KTOT;
    const int khw = r >> 6;
    const int ci  = r & 63;
    const int pw  = widx[o * KTOT + ci * 9 + khw];
    A_lds[o * ASTR + r] = f2bfu(lut_s[pw & 255]);
  }

  // ---- prologue: stage chunk 0 into buffer 0 ----
  {
    const float* xp = chunk_ptr(x, bn, oh, ow0, scol, sg, 0);
    unsigned int pk[8];
#pragma unroll
    for (int j = 0; j < 8; ++j)
      pk[j] = pk_bf16(xp[(size_t)(2 * j) * (H * W)],
                      xp[(size_t)(2 * j + 1) * (H * W)]);
    unsigned int* bp = (unsigned int*)&B_lds[scol * BSTR + sg * 16];
    *(v4ui*)bp       = (v4ui){pk[0], pk[1], pk[2], pk[3]};
    *(v4ui*)(bp + 4) = (v4ui){pk[4], pk[5], pk[6], pk[7]};
  }

  v8f acc[4];
  const v8f vzero = {0.f, 0.f, 0.f, 0.f, 0.f, 0.f, 0.f, 0.f};
#pragma unroll
  for (int nt = 0; nt < 4; ++nt) acc[nt] = vzero;

  __syncthreads();

  // ---- main loop: one barrier per chunk, loads for k+1 overlap compute(k) ----
#pragma unroll 2
  for (int k = 0; k < NCHUNK; ++k) {
    // issue global loads for chunk k+1 (in flight during compute of chunk k)
    float f[16];
    if (k + 1 < NCHUNK) {
      const float* xp = chunk_ptr(x, bn, oh, ow0, scol, sg, k + 1);
#pragma unroll
      for (int j = 0; j < 16; ++j) f[j] = xp[(size_t)j * (H * W)];
      if (k + 2 < NCHUNK)
        __builtin_prefetch(chunk_ptr(x, bn, oh, ow0, scol, sg, k + 2), 0, 1);
    }

    // compute chunk k from buffer k&1
    {
      const unsigned short* Bb = &B_lds[(k & 1) * BBUF];
      const int kbase = (k >> 1) * 64 + (k & 1) * 32;
      const unsigned short* ap = &A_lds[lm * ASTR + kbase + half * 16];
      const v16bf afrag = ld_frag(ap, ap + 8);    // K = half*16+{0..15}
#pragma unroll
      for (int nt = 0; nt < 4; ++nt) {
        const unsigned short* bq = &Bb[(wg * 64 + nt * 16 + lm) * BSTR + half * 16];
        const v16bf bfrag = ld_frag(bq, bq + 8);  // K = half*16+{0..15}
        acc[nt] = __builtin_amdgcn_wmma_f32_16x16x32_bf16(
            false, afrag, false, bfrag, (short)0, acc[nt], false, false);
      }
    }

    // convert + store chunk k+1 into the other buffer
    if (k + 1 < NCHUNK) {
      unsigned int pk[8];
#pragma unroll
      for (int j = 0; j < 8; ++j) pk[j] = pk_bf16(f[2 * j], f[2 * j + 1]);
      unsigned int* bp =
          (unsigned int*)&B_lds[((k + 1) & 1) * BBUF + scol * BSTR + sg * 16];
      *(v4ui*)bp       = (v4ui){pk[0], pk[1], pk[2], pk[3]};
      *(v4ui*)(bp + 4) = (v4ui){pk[4], pk[5], pk[6], pk[7]};
    }

    __syncthreads();   // chunk k+1 ready; everyone done reading buffer k&1
  }

  // ---- epilogue: bias + store. C/D: VGPR r -> M = half*8 + r, N = lane&15 ----
#pragma unroll
  for (int nt = 0; nt < 4; ++nt) {
    const int ow = ow0 + wg * 64 + nt * 16 + lm;
    if (ow < WO) {
#pragma unroll
      for (int r = 0; r < 8; ++r) {
        const int co = mt * 16 + half * 8 + r;
        out[((size_t)(bn * COUT + co) * HO + (size_t)oh) * WO + ow] =
            acc[nt][r] + bias_s[co];
      }
    }
  }
}

extern "C" void kernel_launch(void* const* d_in, const int* in_sizes, int n_in,
                              void* d_out, int out_size, void* d_ws, size_t ws_size,
                              hipStream_t stream) {
  (void)in_sizes; (void)n_in; (void)d_ws; (void)ws_size; (void)out_size;
  const float* x    = (const float*)d_in[0];
  const float* lut  = (const float*)d_in[1];
  const int*   widx = (const int*)d_in[2];
  const float* bias = (const float*)d_in[3];
  float* out = (float*)d_out;

  dim3 grid(NB * HO * 2);   // 16 * 254 * 2 = 8128 blocks
  dim3 block(256);          // 8 wave32 waves
  palet_conv2d_wmma_bf16<<<grid, block, 0, stream>>>(x, lut, widx, bias, out);
}
